// TDTFLayer_22488448761977
// MI455X (gfx1250) — compile-verified, matrix operand order checked by reading, not run on hardware
//
#include <hip/hip_runtime.h>
#include <math.h>

// ---------------------------------------------------------------------------
// Qwen2-ish decoder layer for MI455X (gfx1250, wave32)
//   - all GEMMs through v_wmma_f32_16x16x32_bf16 (fp32 accumulate)
//   - weights pre-transposed to [N][K] bf16 so B fragments are contiguous-K
//   - B tiles staged into LDS by the Tensor Data Mover (tensor_load_to_lds),
//     double-buffered and synchronized with s_wait_tensorcnt
// ---------------------------------------------------------------------------

typedef __bf16 bf16_t;
typedef __attribute__((ext_vector_type(16))) __bf16        v16bf;
typedef __attribute__((ext_vector_type(8)))  __bf16        v8bf;
typedef __attribute__((ext_vector_type(8)))  float         v8f;
typedef __attribute__((ext_vector_type(4)))  unsigned int  v4u;
typedef __attribute__((ext_vector_type(8)))  int           v8i;
typedef __attribute__((ext_vector_type(4)))  int           v4i;
typedef unsigned int u32;

constexpr int  BQ   = 4;
constexpr int  TQ   = 2048;
constexpr int  DQ   = 2048;
constexpr int  FFQ  = 5632;
constexpr int  NTOK = BQ * TQ;           // 8192 tokens
constexpr float EPSQ = 1e-6f;

constexpr int TILE_ELEMS = 64 * 32;      // one B tile: 64 n-rows x 32 k (bf16)
constexpr int TILE_BYTES = TILE_ELEMS * 2;

// ---------------------------------------------------------------------------
// TDM: 2-D tile load (64 rows x 32 elems, bf16) from row-major [rows][K]
// global tensor into contiguous LDS. D# packing per CDNA5 ISA ch.8.
// Issued by one wave; tracked with TENSORcnt.
// ---------------------------------------------------------------------------
__device__ __forceinline__ void tdm_load_tile(const bf16_t* gsrc, u32 lds_off,
                                              int K, int Nrows) {
  unsigned long long ga = (unsigned long long)(uintptr_t)gsrc;
  v4u g0;
  g0[0] = 1u;                                        // count=1 valid descriptor
  g0[1] = lds_off;                                   // LDS byte address
  g0[2] = (u32)ga;                                   // global_addr[31:0]
  g0[3] = (u32)((ga >> 32) & 0x1FFFFFFull)           // global_addr[56:32]
          | (2u << 30);                              // type=2 ("image")
  v8i g1;
  g1[0] = (int)(1u << 16);                           // data_size=1 (2 bytes)
  g1[1] = (int)((u32)K << 16);                       // tensor_dim0[15:0]
  g1[2] = (int)(((u32)K >> 16) | ((u32)Nrows << 16));// dim0[31:16]|dim1[15:0]
  g1[3] = (int)(((u32)Nrows >> 16) | (32u << 16));   // dim1[31:16]|tile_dim0=32
  g1[4] = (int)64u;                                  // tile_dim1=64, tile_dim2=0
  g1[5] = (int)(u32)K;                               // tensor_dim0_stride[31:0]
  g1[6] = 0;                                         // stride0[47:32]|stride1 lo
  g1[7] = 0;
  v4i g2 = {0, 0, 0, 0};
  v4i g3 = {0, 0, 0, 0};
  v8i g4 = {0, 0, 0, 0, 0, 0, 0, 0};                 // unused trailing group
  __builtin_amdgcn_tensor_load_to_lds(g0, g1, g2, g3, g4, 0);
}

// ---------------------------------------------------------------------------
// fp32 [R][C] -> bf16 transposed [C][R], tiled via LDS (one-time weight prep)
// ---------------------------------------------------------------------------
__global__ __launch_bounds__(256)
void cvt_transpose_bf16(const float* __restrict__ in, bf16_t* __restrict__ out,
                        int R, int C) {
  __shared__ float tile[32][33];
  const int c0 = blockIdx.x * 32;
  const int r0 = blockIdx.y * 32;
  const int tx = threadIdx.x & 31;
  const int ty = threadIdx.x >> 5;   // 0..7
#pragma unroll
  for (int i = ty; i < 32; i += 8)
    tile[i][tx] = in[(size_t)(r0 + i) * C + (c0 + tx)];
  __syncthreads();
#pragma unroll
  for (int i = ty; i < 32; i += 8)
    out[(size_t)(c0 + i) * R + (r0 + tx)] = (bf16_t)tile[tx][i];
}

// ---------------------------------------------------------------------------
// Router score + RMSNorm1 fused: one 256-thread block per token
// ---------------------------------------------------------------------------
__global__ __launch_bounds__(256)
void router_norm1(const float* __restrict__ x,
                  const float* __restrict__ w_router,
                  const float* __restrict__ b_router,
                  const float* __restrict__ norm_w,
                  bf16_t* __restrict__ h1,
                  float* __restrict__ mask) {
  const int t = blockIdx.x;
  const float* row = x + (size_t)t * DQ;

  float dot = 0.f, ss = 0.f;
  for (int i = threadIdx.x; i < DQ; i += 256) {
    float v = row[i];
    dot += v * w_router[i];
    ss  += v * v;
  }
  __shared__ float sdot[256];
  __shared__ float sss[256];
  sdot[threadIdx.x] = dot;
  sss[threadIdx.x]  = ss;
  __syncthreads();
  for (int s = 128; s > 0; s >>= 1) {
    if (threadIdx.x < s) {
      sdot[threadIdx.x] += sdot[threadIdx.x + s];
      sss[threadIdx.x]  += sss[threadIdx.x + s];
    }
    __syncthreads();
  }
  __shared__ float inv_s;
  if (threadIdx.x == 0) {
    float score = sdot[0] + b_router[0];
    mask[t] = (score > 0.f) ? 1.f : 0.f;          // sigmoid(s)>0.5 <=> s>0
    inv_s = rsqrtf(sss[0] * (1.f / DQ) + EPSQ);
  }
  __syncthreads();
  const float inv = inv_s;
  for (int i = threadIdx.x; i < DQ; i += 256)
    h1[(size_t)t * DQ + i] = (bf16_t)(row[i] * inv * norm_w[i]);
}

// RMSNorm only (for h -> h2)
__global__ __launch_bounds__(256)
void rmsnorm_only(const float* __restrict__ h,
                  const float* __restrict__ norm_w,
                  bf16_t* __restrict__ h2) {
  const int t = blockIdx.x;
  const float* row = h + (size_t)t * DQ;
  float ss = 0.f;
  for (int i = threadIdx.x; i < DQ; i += 256) {
    float v = row[i];
    ss += v * v;
  }
  __shared__ float sss[256];
  sss[threadIdx.x] = ss;
  __syncthreads();
  for (int s = 128; s > 0; s >>= 1) {
    if (threadIdx.x < s) sss[threadIdx.x] += sss[threadIdx.x + s];
    __syncthreads();
  }
  __shared__ float inv_s;
  if (threadIdx.x == 0) inv_s = rsqrtf(sss[0] * (1.f / DQ) + EPSQ);
  __syncthreads();
  const float inv = inv_s;
  for (int i = threadIdx.x; i < DQ; i += 256)
    h2[(size_t)t * DQ + i] = (bf16_t)(row[i] * inv * norm_w[i]);
}

// ---------------------------------------------------------------------------
// WMMA bf16 GEMM: C[M,N] = A[M,K] (row-major bf16) @ W (given as Wt[N][K] bf16)
//   block = 256 threads (8 waves); block tile 128(M) x 64(N)
//   wave w owns rows [bm+16w, bm+16w+16), all 64 N cols -> 4 v8f accumulators
//   B tiles (64 n-rows x 32 k) staged into LDS by the TDM, double buffered.
//   MODE 0: out_bf16 = C + bias[col]                       (v = h1@wv + bv)
//   MODE 1: out_f32  = res + C                             (h = x + v@wo)
//   MODE 2: out_bf16 = silu(C0) * C1  (two weight streams) (SwiGLU activation)
//   MODE 3: out_f32  = mask[row] ? res + C : passx         (final output)
// ---------------------------------------------------------------------------
__device__ __forceinline__ v16bf make_frag(v8bf lo, v8bf hi) {
  v16bf f;
#pragma unroll
  for (int i = 0; i < 8; ++i) { f[i] = lo[i]; f[8 + i] = hi[i]; }
  return f;
}

template <int MODE>
__global__ __launch_bounds__(256)
void gemm_wmma(const bf16_t* __restrict__ A,
               const bf16_t* __restrict__ W0t,   // [N][K]
               const bf16_t* __restrict__ W1t,   // [N][K] (MODE 2)
               const float*  __restrict__ bias,
               const float*  __restrict__ res,
               const float*  __restrict__ passx,
               const float*  __restrict__ mask,
               void*         __restrict__ outv,
               int M, int N, int K) {
  constexpr int NT = (MODE == 2) ? 2 : 1;            // weight streams
  __shared__ bf16_t lds[2 * NT * TILE_ELEMS];        // 2 buffers x NT tiles
  const u32 lds_base = (u32)(uintptr_t)(&lds[0]);

  const int lane = threadIdx.x & 31;
  const int wave = threadIdx.x >> 5;
  const int bn   = blockIdx.x * 64;
  const int bm   = blockIdx.y * 128;

  v8f acc0[4] = {};
  v8f acc1[4] = {};
  (void)acc1;

  // A fragment addressing per 16-bit A 16x32 layout:
  // lane<16 holds K {0..7,16..23}; lane>=16 holds K {8..15,24..31}.
  const int mrow = bm + wave * 16 + (lane & 15);
  const int kb   = (lane < 16) ? 0 : 8;
  const bf16_t* aptr = A + (size_t)mrow * K + kb;

  const int nsteps = K / 32;
  const int issuer = (threadIdx.x < 32);             // wave 0 drives the TDM

  if (issuer) {
    tdm_load_tile(W0t + (size_t)bn * K, lds_base, K, N);
    if (MODE == 2)
      tdm_load_tile(W1t + (size_t)bn * K, lds_base + TILE_BYTES, K, N);
  }

  int buf = 0;
  for (int step = 0; step < nsteps; ++step) {
    const int k0 = step * 32;
    __syncthreads();                 // prior reads of buf^1 complete
    if (issuer) {
      if (step + 1 < nsteps) {
        const u32 dst = lds_base + (u32)((buf ^ 1) * NT) * TILE_BYTES;
        tdm_load_tile(W0t + (size_t)bn * K + (k0 + 32), dst, K, N);
        if (MODE == 2)
          tdm_load_tile(W1t + (size_t)bn * K + (k0 + 32), dst + TILE_BYTES,
                        K, N);
        __builtin_amdgcn_s_wait_tensorcnt((short)NT);  // tile(step) landed
      } else {
        __builtin_amdgcn_s_wait_tensorcnt((short)0);
      }
    }
    __syncthreads();                 // buf ready for all waves

    // A fragment straight from global: two 16B loads per lane
    v8bf alo = *(const v8bf*)(aptr + k0);
    v8bf ahi = *(const v8bf*)(aptr + k0 + 16);
    if (k0 + 32 < K) __builtin_prefetch(aptr + k0 + 32, 0, 1);
    const v16bf afrag = make_frag(alo, ahi);

    // batch all B fragments, then run the WMMA chain back-to-back
    const bf16_t* t0 = lds + (buf * NT) * TILE_ELEMS;
    v16bf bfrag[4];
#pragma unroll
    for (int j = 0; j < 4; ++j) {
      const bf16_t* bcol = t0 + (j * 16 + (lane & 15)) * 32 + kb;
      bfrag[j] = make_frag(*(const v8bf*)bcol, *(const v8bf*)(bcol + 16));
    }
#pragma unroll
    for (int j = 0; j < 4; ++j)
      acc0[j] = __builtin_amdgcn_wmma_f32_16x16x32_bf16(
          false, afrag, false, bfrag[j], (short)0, acc0[j], false, false);

    if (MODE == 2) {
      const bf16_t* t1 = t0 + TILE_ELEMS;
#pragma unroll
      for (int j = 0; j < 4; ++j) {
        const bf16_t* bcol = t1 + (j * 16 + (lane & 15)) * 32 + kb;
        bfrag[j] = make_frag(*(const v8bf*)bcol, *(const v8bf*)(bcol + 16));
      }
#pragma unroll
      for (int j = 0; j < 4; ++j)
        acc1[j] = __builtin_amdgcn_wmma_f32_16x16x32_bf16(
            false, afrag, false, bfrag[j], (short)0, acc1[j], false, false);
    }
    buf ^= 1;
  }

  // C/D layout: lanes 0-15 -> N=lane, M=vgpr; lanes 16-31 -> N=lane-16, M=8+vgpr
  const int rbase = (lane < 16) ? 0 : 8;
  const int coll  = lane & 15;
#pragma unroll
  for (int j = 0; j < 4; ++j) {
    const int col = bn + j * 16 + coll;
#pragma unroll
    for (int i = 0; i < 8; ++i) {
      const int row = bm + wave * 16 + rbase + i;
      const size_t idx = (size_t)row * N + col;
      const float a = acc0[j][i];
      if (MODE == 0) {
        ((bf16_t*)outv)[idx] = (bf16_t)(a + bias[col]);
      } else if (MODE == 1) {
        ((float*)outv)[idx] = res[idx] + a;
      } else if (MODE == 2) {
        const float g = a, u = acc1[j][i];
        const float s = g / (1.f + __expf(-g));  // silu
        ((bf16_t*)outv)[idx] = (bf16_t)(s * u);
      } else {
        const float p = res[idx] + a;
        ((float*)outv)[idx] = (mask[row] > 0.5f) ? p : passx[idx];
      }
    }
  }
}

// ---------------------------------------------------------------------------
// Host-side orchestration
// ---------------------------------------------------------------------------
extern "C" void kernel_launch(void* const* d_in, const int* in_sizes, int n_in,
                              void* d_out, int out_size, void* d_ws,
                              size_t ws_size, hipStream_t stream) {
  const float* x        = (const float*)d_in[0];   // [B,T,D]
  const float* w_router = (const float*)d_in[1];   // [D]
  const float* b_router = (const float*)d_in[2];   // scalar
  const float* norm1_w  = (const float*)d_in[3];   // [D]
  const float* wv       = (const float*)d_in[4];   // [D,D]
  const float* bv       = (const float*)d_in[5];   // [D]
  const float* wo       = (const float*)d_in[6];   // [D,D]
  const float* norm2_w  = (const float*)d_in[7];   // [D]
  const float* w_gate   = (const float*)d_in[8];   // [D,FF]
  const float* w_up     = (const float*)d_in[9];   // [D,FF]
  const float* w_down   = (const float*)d_in[10];  // [FF,D]
  (void)in_sizes; (void)n_in; (void)out_size; (void)ws_size;

  // workspace carve-out (256B aligned)
  char* ws = (char*)d_ws;
  size_t off = 0;
  auto carve = [&](size_t bytes) -> char* {
    char* p = ws + off;
    off += (bytes + 255) & ~(size_t)255;
    return p;
  };
  bf16_t* wv_t  = (bf16_t*)carve((size_t)DQ * DQ * sizeof(bf16_t));   // [D][D]
  bf16_t* wo_t  = (bf16_t*)carve((size_t)DQ * DQ * sizeof(bf16_t));   // [D][D]
  bf16_t* wg_t  = (bf16_t*)carve((size_t)FFQ * DQ * sizeof(bf16_t));  // [FF][D]
  bf16_t* wu_t  = (bf16_t*)carve((size_t)FFQ * DQ * sizeof(bf16_t));  // [FF][D]
  bf16_t* wd_t  = (bf16_t*)carve((size_t)DQ * FFQ * sizeof(bf16_t));  // [D][FF]
  bf16_t* h1_b  = (bf16_t*)carve((size_t)NTOK * DQ * sizeof(bf16_t));
  bf16_t* v_b   = (bf16_t*)carve((size_t)NTOK * DQ * sizeof(bf16_t));
  bf16_t* h2_b  = (bf16_t*)carve((size_t)NTOK * DQ * sizeof(bf16_t));
  bf16_t* act_b = (bf16_t*)carve((size_t)NTOK * FFQ * sizeof(bf16_t));
  float*  h_f   = (float*)carve((size_t)NTOK * DQ * sizeof(float));
  float*  mask  = (float*)carve((size_t)NTOK * sizeof(float));

  // 1) weight fp32 [K][N] -> bf16 transposed [N][K]
  cvt_transpose_bf16<<<dim3(DQ / 32, DQ / 32), 256, 0, stream>>>(
      wv, wv_t, DQ, DQ);
  cvt_transpose_bf16<<<dim3(DQ / 32, DQ / 32), 256, 0, stream>>>(
      wo, wo_t, DQ, DQ);
  cvt_transpose_bf16<<<dim3(FFQ / 32, DQ / 32), 256, 0, stream>>>(
      w_gate, wg_t, DQ, FFQ);
  cvt_transpose_bf16<<<dim3(FFQ / 32, DQ / 32), 256, 0, stream>>>(
      w_up, wu_t, DQ, FFQ);
  cvt_transpose_bf16<<<dim3(DQ / 32, FFQ / 32), 256, 0, stream>>>(
      w_down, wd_t, FFQ, DQ);

  // 2) router + rmsnorm1
  router_norm1<<<NTOK, 256, 0, stream>>>(x, w_router, b_router, norm1_w,
                                         h1_b, mask);

  // 3) v = h1 @ wv + bv         (M=8192, N=2048, K=2048)
  gemm_wmma<0><<<dim3(DQ / 64, NTOK / 128), 256, 0, stream>>>(
      h1_b, wv_t, nullptr, bv, nullptr, nullptr, nullptr, (void*)v_b,
      NTOK, DQ, DQ);

  // 4) h = x + v @ wo           (M=8192, N=2048, K=2048)
  gemm_wmma<1><<<dim3(DQ / 64, NTOK / 128), 256, 0, stream>>>(
      v_b, wo_t, nullptr, nullptr, x, nullptr, nullptr, (void*)h_f,
      NTOK, DQ, DQ);

  // 5) h2 = rmsnorm(h)
  rmsnorm_only<<<NTOK, 256, 0, stream>>>(h_f, norm2_w, h2_b);

  // 6) act = silu(h2@w_gate) * (h2@w_up)   (M=8192, N=5632, K=2048)
  gemm_wmma<2><<<dim3(FFQ / 64, NTOK / 128), 256, 0, stream>>>(
      h2_b, wg_t, wu_t, nullptr, nullptr, nullptr, nullptr, (void*)act_b,
      NTOK, FFQ, DQ);

  // 7) out = routed ? h + act@w_down : x   (M=8192, N=2048, K=5632)
  gemm_wmma<3><<<dim3(DQ / 64, NTOK / 128), 256, 0, stream>>>(
      act_b, wd_t, nullptr, nullptr, h_f, x, mask, d_out,
      NTOK, DQ, FFQ);
}
